// ShuffleSelfAttention_70626442215937
// MI455X (gfx1250) — matrix operand order repaired and checked by reading, not compiled, not run on hardware
//
#include <hip/hip_runtime.h>

// ---------------------------------------------------------------------------
// Problem constants (from reference): B=4, D=2048, F=64, H=8, C=V=2
// ---------------------------------------------------------------------------
#define FDIM    64
#define DDEPTH  2048
#define NHEAD   8
#define NBATCH  4
#define HB      32          // NHEAD * NBATCH
#define DH      128         // per-head channels (C*F)
#define QKOUT   1024        // H * C * F
#define OUT_ELEMS 8388608   // B * 16 * D * F  (first output), attn follows

typedef __attribute__((ext_vector_type(16))) __bf16 v16bf;
typedef __attribute__((ext_vector_type(8)))  __bf16 v8bf;
typedef __attribute__((ext_vector_type(8)))  float  v8f;

// ---------------------------------------------------------------------------
// Fragment helpers (wave32 WMMA layouts, CDNA5 ISA §7.12.2)
// A 16x32 bf16: lane L -> row M=L%16; 8 elems at K0=(L/16)*8, 8 at K0+16
// B 32x16 bf16: lane L -> col N=L%16; 8 elems at K0=(L/16)*8, 8 at K0+16
// C/D 16x16 f32: lane L -> col N=L%16, rows M=(L/16)*8 + r
// ---------------------------------------------------------------------------
static __device__ __forceinline__ v16bf pack16(v8bf lo, v8bf hi) {
    v16bf r;
#pragma unroll
    for (int i = 0; i < 8; ++i) { r[i] = lo[i]; r[i + 8] = hi[i]; }
    return r;
}
static __device__ __forceinline__ v16bf load_a_bf16(const __bf16* p) {
    return pack16(*(const v8bf*)p, *(const v8bf*)(p + 16));
}
static __device__ __forceinline__ v16bf load_b_bf16(const __bf16* p) {
    return pack16(*(const v8bf*)p, *(const v8bf*)(p + 16));
}
static __device__ __forceinline__ v16bf load_a_f32(const float* p) {
    v16bf r;
#pragma unroll
    for (int i = 0; i < 8; ++i) r[i] = (__bf16)p[i];
#pragma unroll
    for (int i = 0; i < 8; ++i) r[i + 8] = (__bf16)p[16 + i];
    return r;
}
static __device__ __forceinline__ v8f wmma_bf16(v16bf a, v16bf b, v8f c) {
    return __builtin_amdgcn_wmma_f32_16x16x32_bf16(false, a, false, b,
                                                   (short)0, c, false, false);
}

// ---------------------------------------------------------------------------
// Kernel 0: W[64][1024] fp32 -> WT[1024][64] bf16, for all three matrices
// ---------------------------------------------------------------------------
__global__ __launch_bounds__(256) void k_transpose_w(
    const float* __restrict__ Wq, const float* __restrict__ Wk,
    const float* __restrict__ Wv, __bf16* __restrict__ WT)
{
    int idx = blockIdx.x * 256 + threadIdx.x;      // 3*64*1024 total
    int mtx = idx / (FDIM * QKOUT);
    int rem = idx % (FDIM * QKOUT);
    int kk  = rem / QKOUT;
    int n   = rem % QKOUT;
    const float* W = (mtx == 0) ? Wq : (mtx == 1) ? Wk : Wv;
    WT[(size_t)mtx * QKOUT * FDIM + (size_t)n * FDIM + kk] = (__bf16)W[rem];
}

// ---------------------------------------------------------------------------
// Kernel 1: QKV projection body. feat[8192][64] x WT(+bias) -> bf16.
//   IS_V=false: head-split  [HB][D][DH]   (scalar b16 stores, stride DH)
//   IS_V=true : transposed  [HB][DH][D]   (one contiguous v8bf store)
// Block: 256 thr = 8 waves; block = 16 rows x 512 cols; wave = 16x64.
// Layout is a compile-time property -> fully branchless epilogue.
// ---------------------------------------------------------------------------
template <bool IS_V>
static __device__ __forceinline__ void proj_body(
    const float* __restrict__ feat, const float* __restrict__ bias,
    const __bf16* __restrict__ wt, __bf16* __restrict__ dst,
    int m0, int n0)
{
    const int lane = threadIdx.x & 31;
    const int lg = lane >> 4, lm = lane & 15;

    const int batch = m0 >> 11;                    // tile-constant batch
    const int d0    = (m0 & 2047) + lg * 8;        // first depth row, this lane

    // A fragments: one feature row per lane, K=64 -> 2 frags
    const float* fr = feat + (size_t)(m0 + lm) * FDIM;
    v16bf a0 = load_a_f32(fr + 0  + lg * 8);
    v16bf a1 = load_a_f32(fr + 32 + lg * 8);

#pragma unroll
    for (int nt = 0; nt < 4; ++nt) {
        const int ncol = n0 + nt * 16 + lm;        // this lane's output column
        const __bf16* wr = wt + (size_t)ncol * FDIM;
        v8f acc = {0.f, 0.f, 0.f, 0.f, 0.f, 0.f, 0.f, 0.f};
        acc = wmma_bf16(a0, load_b_bf16(wr + 0  + lg * 8), acc);
        acc = wmma_bf16(a1, load_b_bf16(wr + 32 + lg * 8), acc);

        const float bterm = bias[ncol];
        const int h  = ncol >> 7;                  // head
        const int c  = ncol & 127;                 // channel within head
        const int bh = h * NBATCH + batch;         // split-heads batch-head

        if (IS_V) {
            // v^T layout: rows r are contiguous -> single 16B store
            v8bf pk;
#pragma unroll
            for (int r = 0; r < 8; ++r) pk[r] = (__bf16)(acc[r] + bterm);
            *(v8bf*)(dst + (size_t)(bh * DH + c) * DDEPTH + d0) = pk;
        } else {
            __bf16* dp = dst + (size_t)(bh * DDEPTH + d0) * DH + c;
#pragma unroll
            for (int r = 0; r < 8; ++r)
                dp[(size_t)r * DH] = (__bf16)(acc[r] + bterm);
        }
    }
}

__global__ __launch_bounds__(256) void k_proj_qk(
    const float* __restrict__ feat,
    const float* __restrict__ bq, const float* __restrict__ bk,
    const __bf16* __restrict__ WT,
    __bf16* __restrict__ qws, __bf16* __restrict__ kws)
{
    const int wid = threadIdx.x >> 5;
    const int mtx = blockIdx.z;                    // 0=q, 1=k (uniform select)
    proj_body<false>(feat,
                     (mtx == 0) ? bq : bk,
                     WT + (size_t)mtx * QKOUT * FDIM,
                     (mtx == 0) ? qws : kws,
                     blockIdx.y * 16, blockIdx.x * 512 + wid * 64);
}

__global__ __launch_bounds__(256) void k_proj_v(
    const float* __restrict__ feat, const float* __restrict__ bv,
    const __bf16* __restrict__ WT, __bf16* __restrict__ vtws)
{
    const int wid = threadIdx.x >> 5;
    proj_body<true>(feat, bv, WT + (size_t)2 * QKOUT * FDIM, vtws,
                    blockIdx.y * 16, blockIdx.x * 512 + wid * 64);
}

// ---------------------------------------------------------------------------
// Kernel 2: logits S = (Q K^T) / sqrt(D), fp32 -> attn region of d_out.
// Grid (16,16,32): 128x128 tile per block; waves 4(m) x 2(n): 32x64 each.
// ---------------------------------------------------------------------------
__global__ __launch_bounds__(256) void k_qk(
    const __bf16* __restrict__ q, const __bf16* __restrict__ k,
    float* __restrict__ attn)
{
    const int lane = threadIdx.x & 31;
    const int wid  = threadIdx.x >> 5;
    const int lg = lane >> 4, lm = lane & 15;
    const int bh = blockIdx.z;
    const int m0 = blockIdx.y * 128 + (wid & 3) * 32;
    const int n0 = blockIdx.x * 128 + (wid >> 2) * 64;

    // hoisted lane-base pointers; k-loop steps them by +32 elements
    const __bf16* qp0 = q + (size_t)bh * DDEPTH * DH
                          + (size_t)(m0 + lm) * DH + lg * 8;
    const __bf16* qp1 = qp0 + 16 * DH;
    const __bf16* kp[4];
#pragma unroll
    for (int nt = 0; nt < 4; ++nt)
        kp[nt] = k + (size_t)bh * DDEPTH * DH
                   + (size_t)(n0 + nt * 16 + lm) * DH + lg * 8;

    v8f acc[2][4];
#pragma unroll
    for (int i = 0; i < 2; ++i)
#pragma unroll
        for (int j = 0; j < 4; ++j)
            acc[i][j] = (v8f){0.f, 0.f, 0.f, 0.f, 0.f, 0.f, 0.f, 0.f};

#pragma unroll
    for (int kk = 0; kk < DH; kk += 32) {
        v16bf a0 = load_a_bf16(qp0 + kk);
        v16bf a1 = load_a_bf16(qp1 + kk);
        v16bf bb[4];
#pragma unroll
        for (int nt = 0; nt < 4; ++nt)
            bb[nt] = load_b_bf16(kp[nt] + kk);
#pragma unroll
        for (int nt = 0; nt < 4; ++nt) {
            acc[0][nt] = wmma_bf16(a0, bb[nt], acc[0][nt]);
            acc[1][nt] = wmma_bf16(a1, bb[nt], acc[1][nt]);
        }
    }

    const float scale = 0.0220970869120796f;       // 1/sqrt(2048)
    float* arow = attn + ((size_t)bh * DDEPTH + m0 + lg * 8) * DDEPTH + n0 + lm;
#pragma unroll
    for (int mt = 0; mt < 2; ++mt)
#pragma unroll
        for (int nt = 0; nt < 4; ++nt) {
            float* ap = arow + (size_t)mt * 16 * DDEPTH + nt * 16;
#pragma unroll
            for (int r = 0; r < 8; ++r)
                ap[(size_t)r * DDEPTH] = acc[mt][nt][r] * scale;
        }
}

// ---------------------------------------------------------------------------
// Kernel 3: in-place row softmax over attn [HB*D rows of 2048].
// One 256-thread block per row, 8 elems/thread, LDS tree reductions.
// ---------------------------------------------------------------------------
__global__ __launch_bounds__(256) void k_softmax(float* __restrict__ attn)
{
    __shared__ float red[256];
    float* p = attn + (size_t)blockIdx.x * DDEPTH;
    const int t = threadIdx.x;

    float4 x0 = ((const float4*)p)[t * 2];
    float4 x1 = ((const float4*)p)[t * 2 + 1];

    float m = fmaxf(fmaxf(fmaxf(x0.x, x0.y), fmaxf(x0.z, x0.w)),
                    fmaxf(fmaxf(x1.x, x1.y), fmaxf(x1.z, x1.w)));
    red[t] = m;
    __syncthreads();
#pragma unroll
    for (int s = 128; s > 0; s >>= 1) {
        if (t < s) red[t] = fmaxf(red[t], red[t + s]);
        __syncthreads();
    }
    const float rmax = red[0];
    __syncthreads();

    x0.x = __expf(x0.x - rmax); x0.y = __expf(x0.y - rmax);
    x0.z = __expf(x0.z - rmax); x0.w = __expf(x0.w - rmax);
    x1.x = __expf(x1.x - rmax); x1.y = __expf(x1.y - rmax);
    x1.z = __expf(x1.z - rmax); x1.w = __expf(x1.w - rmax);

    red[t] = x0.x + x0.y + x0.z + x0.w + x1.x + x1.y + x1.z + x1.w;
    __syncthreads();
#pragma unroll
    for (int s = 128; s > 0; s >>= 1) {
        if (t < s) red[t] += red[t + s];
        __syncthreads();
    }
    const float inv = 1.0f / red[0];

    x0.x *= inv; x0.y *= inv; x0.z *= inv; x0.w *= inv;
    x1.x *= inv; x1.y *= inv; x1.z *= inv; x1.w *= inv;
    ((float4*)p)[t * 2]     = x0;
    ((float4*)p)[t * 2 + 1] = x1;
}

// ---------------------------------------------------------------------------
// Kernel 4: out = P @ V.  P fp32 (attn) converted on the fly; V^T bf16.
// Grid (16,32): block = 128 rows x full 128 cols; waves 4(m) x 2(n).
// K-loop unrolled x2 (16 WMMA/trip) so cvt VALU co-executes with WMMA.
// Store into merged-head layout [B][16][D][64].
// ---------------------------------------------------------------------------
__global__ __launch_bounds__(256) void k_pv(
    const float* __restrict__ attn, const __bf16* __restrict__ vt,
    float* __restrict__ out)
{
    const int lane = threadIdx.x & 31;
    const int wid  = threadIdx.x >> 5;
    const int lg = lane >> 4, lm = lane & 15;
    const int bh = blockIdx.y;
    const int m0 = blockIdx.x * 128 + (wid & 3) * 32;
    const int n0 = (wid >> 2) * 64;

    // hoisted lane-base pointers; k-loop steps by +32 elements
    const float* pp0 = attn + (size_t)bh * DDEPTH * DDEPTH
                            + (size_t)(m0 + lm) * DDEPTH + lg * 8;
    const float* pp1 = pp0 + 16 * DDEPTH;
    const __bf16* vp[4];
#pragma unroll
    for (int nt = 0; nt < 4; ++nt)
        vp[nt] = vt + (size_t)bh * DH * DDEPTH
                    + (size_t)(n0 + nt * 16 + lm) * DDEPTH + lg * 8;

    v8f acc[2][4];
#pragma unroll
    for (int i = 0; i < 2; ++i)
#pragma unroll
        for (int j = 0; j < 4; ++j)
            acc[i][j] = (v8f){0.f, 0.f, 0.f, 0.f, 0.f, 0.f, 0.f, 0.f};

#pragma unroll 2
    for (int kk = 0; kk < DDEPTH; kk += 32) {
        v16bf a0 = load_a_f32(pp0 + kk);
        v16bf a1 = load_a_f32(pp1 + kk);
        v16bf bb[4];
#pragma unroll
        for (int nt = 0; nt < 4; ++nt)
            bb[nt] = load_b_bf16(vp[nt] + kk);
#pragma unroll
        for (int nt = 0; nt < 4; ++nt) {
            acc[0][nt] = wmma_bf16(a0, bb[nt], acc[0][nt]);
            acc[1][nt] = wmma_bf16(a1, bb[nt], acc[1][nt]);
        }
    }

    const int h = bh >> 2;                         // bh = h*NBATCH + b
    const int b = bh & 3;
#pragma unroll
    for (int mt = 0; mt < 2; ++mt)
#pragma unroll
        for (int nt = 0; nt < 4; ++nt) {
            const int j  = n0 + nt * 16 + lm;      // 0..127
            const int vc = j >> 6;                 // v-channel 0/1
            const int f  = j & 63;
            float* op = out + ((size_t)(b * 16 + h * 2 + vc) * DDEPTH
                               + m0 + mt * 16 + lg * 8) * FDIM + f;
#pragma unroll
            for (int r = 0; r < 8; ++r)
                op[(size_t)r * FDIM] = acc[mt][nt][r];
        }
}

// ---------------------------------------------------------------------------
// Launcher
// ---------------------------------------------------------------------------
extern "C" void kernel_launch(void* const* d_in, const int* in_sizes, int n_in,
                              void* d_out, int out_size, void* d_ws, size_t ws_size,
                              hipStream_t stream)
{
    const float* feat = (const float*)d_in[0];
    const float* Wq   = (const float*)d_in[1];
    const float* bq   = (const float*)d_in[2];
    const float* Wk   = (const float*)d_in[3];
    const float* bk   = (const float*)d_in[4];
    const float* Wv   = (const float*)d_in[5];
    const float* bv   = (const float*)d_in[6];

    float* out  = (float*)d_out;
    float* attn = out + (size_t)OUT_ELEMS;         // second tuple output

    char* ws = (char*)d_ws;
    __bf16* qws  = (__bf16*)(ws);                                   // 16 MB
    __bf16* kws  = (__bf16*)(ws + (size_t)16 * 1024 * 1024);        // 16 MB
    __bf16* vtws = (__bf16*)(ws + (size_t)32 * 1024 * 1024);        // 16 MB
    __bf16* WT   = (__bf16*)(ws + (size_t)48 * 1024 * 1024);        // 384 KB

    k_transpose_w<<<dim3(768), dim3(256), 0, stream>>>(Wq, Wk, Wv, WT);

    k_proj_qk<<<dim3(2, 512, 2), dim3(256), 0, stream>>>(
        feat, bq, bk, WT, qws, kws);
    k_proj_v<<<dim3(2, 512), dim3(256), 0, stream>>>(
        feat, bv, WT, vtws);

    k_qk<<<dim3(16, 16, 32), dim3(256), 0, stream>>>(qws, kws, attn);

    k_softmax<<<dim3(HB * DDEPTH), dim3(256), 0, stream>>>(attn);

    k_pv<<<dim3(16, 32), dim3(256), 0, stream>>>(attn, vtws, out);
}
